// RootLayer_7962869366906
// MI455X (gfx1250) — matrix operand order repaired and checked by reading, not compiled, not run on hardware
//
#include <hip/hip_runtime.h>
#include <hip/hip_bf16.h>

typedef __attribute__((ext_vector_type(16))) _Float16 v16h;
typedef __attribute__((ext_vector_type(4)))  _Float16 v4h;
typedef __attribute__((ext_vector_type(8)))  float    v8f;
typedef __attribute__((ext_vector_type(4)))  float    v4f;

#define NFEAT  1024   // K
#define NCLS   128    // N
#define NBATCH 2048   // M

#define LOG2E  1.4426950408889634f
#define LN2    0.6931471805599453f

// sched_barrier mask: VALU(0x2)|SALU(0x4) may cross; VMEM and WMMA may NOT.
#define SCHED_FENCE() __builtin_amdgcn_sched_barrier(0x6)

// ---------------- wave32 reduction helpers (ds_swizzle / ds_bpermute) ----------------
template <int XMASK>
__device__ __forceinline__ float swz_xor(float v) {
  return __int_as_float(__builtin_amdgcn_ds_swizzle(__float_as_int(v), (XMASK << 10) | 0x1F));
}
__device__ __forceinline__ float wave_max32(float v) {
  v = fmaxf(v, swz_xor<16>(v));
  v = fmaxf(v, swz_xor<8>(v));
  v = fmaxf(v, swz_xor<4>(v));
  v = fmaxf(v, swz_xor<2>(v));
  v = fmaxf(v, swz_xor<1>(v));
  return v;
}
__device__ __forceinline__ float wave_sum32(float v) {
  v += swz_xor<16>(v);
  v += swz_xor<8>(v);
  v += swz_xor<4>(v);
  v += swz_xor<2>(v);
  v += swz_xor<1>(v);
  return v;
}
__device__ __forceinline__ float bcast_lane(float v, int srcLane) {
  return __int_as_float(__builtin_amdgcn_ds_bpermute(srcLane << 2, __float_as_int(v)));
}
__device__ __forceinline__ float hw_exp2(float v) { return __builtin_amdgcn_exp2f(v); }
__device__ __forceinline__ float hw_log2(float v) { return __builtin_amdgcn_logf(v); }

// ---------------- kernel 1: p = softmax(weight) rows, stored f16 row-major (C,F) ----------------
__global__ void __launch_bounds__(128, 1)
wsoftmax_f16_kernel(const float* __restrict__ w, _Float16* __restrict__ p) {
  const int lane = threadIdx.x & 31;
  const int wv   = threadIdx.x >> 5;
  const int c    = blockIdx.x * (blockDim.x >> 5) + wv;   // one wave per class row
  if (c >= NCLS) return;
  const float* row = w + c * NFEAT;

  v4f v[8];
  float m = -3.402823466e38f;
#pragma unroll
  for (int j = 0; j < 8; ++j) {
    v[j] = *(const v4f*)(row + ((j * 32 + lane) << 2));
    m = fmaxf(m, fmaxf(fmaxf(v[j].x, v[j].y), fmaxf(v[j].z, v[j].w)));
  }
  m = wave_max32(m);

  float s = 0.f;
  const float ms = m * LOG2E;
#pragma unroll
  for (int j = 0; j < 8; ++j) {
    s += hw_exp2(__builtin_fmaf(v[j].x, LOG2E, -ms)) +
         hw_exp2(__builtin_fmaf(v[j].y, LOG2E, -ms)) +
         hw_exp2(__builtin_fmaf(v[j].z, LOG2E, -ms)) +
         hw_exp2(__builtin_fmaf(v[j].w, LOG2E, -ms));
  }
  s = wave_sum32(s);

  const float d2 = ms + hw_log2(s);                       // log2-normalizer
  _Float16* prow = p + c * NFEAT;
#pragma unroll
  for (int j = 0; j < 8; ++j) {
    v4h h;
    h.x = (_Float16)hw_exp2(__builtin_fmaf(v[j].x, LOG2E, -d2));
    h.y = (_Float16)hw_exp2(__builtin_fmaf(v[j].y, LOG2E, -d2));
    h.z = (_Float16)hw_exp2(__builtin_fmaf(v[j].z, LOG2E, -d2));
    h.w = (_Float16)hw_exp2(__builtin_fmaf(v[j].w, LOG2E, -d2));
    *(v4h*)(prow + ((j * 32 + lane) << 2)) = h;
  }
}

// ---------------- kernel 2 helpers ----------------
struct XQ { v4f a, b, c, d; };

__device__ __forceinline__ XQ ldx(const float* xa) {
  XQ q;
  q.a = *(const v4f*)(xa);        // K = +0..3
  q.b = *(const v4f*)(xa + 4);    // K = +4..7
  q.c = *(const v4f*)(xa + 8);    // K = +8..11
  q.d = *(const v4f*)(xa + 12);   // K = +12..15
  return q;
}
__device__ __forceinline__ v16h expa(const XQ& q, float mx2) {
  v16h a;
  a[0]  = (_Float16)hw_exp2(__builtin_fmaf(q.a.x, LOG2E, -mx2));
  a[1]  = (_Float16)hw_exp2(__builtin_fmaf(q.a.y, LOG2E, -mx2));
  a[2]  = (_Float16)hw_exp2(__builtin_fmaf(q.a.z, LOG2E, -mx2));
  a[3]  = (_Float16)hw_exp2(__builtin_fmaf(q.a.w, LOG2E, -mx2));
  a[4]  = (_Float16)hw_exp2(__builtin_fmaf(q.b.x, LOG2E, -mx2));
  a[5]  = (_Float16)hw_exp2(__builtin_fmaf(q.b.y, LOG2E, -mx2));
  a[6]  = (_Float16)hw_exp2(__builtin_fmaf(q.b.z, LOG2E, -mx2));
  a[7]  = (_Float16)hw_exp2(__builtin_fmaf(q.b.w, LOG2E, -mx2));
  a[8]  = (_Float16)hw_exp2(__builtin_fmaf(q.c.x, LOG2E, -mx2));
  a[9]  = (_Float16)hw_exp2(__builtin_fmaf(q.c.y, LOG2E, -mx2));
  a[10] = (_Float16)hw_exp2(__builtin_fmaf(q.c.z, LOG2E, -mx2));
  a[11] = (_Float16)hw_exp2(__builtin_fmaf(q.c.w, LOG2E, -mx2));
  a[12] = (_Float16)hw_exp2(__builtin_fmaf(q.d.x, LOG2E, -mx2));
  a[13] = (_Float16)hw_exp2(__builtin_fmaf(q.d.y, LOG2E, -mx2));
  a[14] = (_Float16)hw_exp2(__builtin_fmaf(q.d.z, LOG2E, -mx2));
  a[15] = (_Float16)hw_exp2(__builtin_fmaf(q.d.w, LOG2E, -mx2));
  return a;
}

#define WMMA16(A, B, C) \
  __builtin_amdgcn_wmma_f32_16x16x32_f16(false, (A), false, (B), (short)0, (C), false, false)

#define LDB4(B0, B1, B2, B3, KC) do {                               \
    B0 = *(const v16h*)(pbase + 0 * (16 * NFEAT) + (KC));           \
    B1 = *(const v16h*)(pbase + 1 * (16 * NFEAT) + (KC));           \
    B2 = *(const v16h*)(pbase + 2 * (16 * NFEAT) + (KC));           \
    B3 = *(const v16h*)(pbase + 3 * (16 * NFEAT) + (KC));           \
  } while (0)

// ---------------- kernel 2: out = mx + ln( exp(x - mx) @ p^T ) via WMMA ----------------
// One wave computes a 16(M) x 64(N) tile: 4 accumulators, double-buffered K loop.
__global__ void __launch_bounds__(128, 1)
lse_wmma_kernel(const float* __restrict__ x,
                const _Float16* __restrict__ p,
                float* __restrict__ out) {
  const int lane = threadIdx.x & 31;
  const int wv   = threadIdx.x >> 5;
  const int gw   = blockIdx.x * (blockDim.x >> 5) + wv;  // global wave id
  const int m0   = (gw >> 1) * 16;                       // 16-row batch tile
  const int nh   = (gw & 1) * 64;                        // which 64-class half
  const int n    = lane & 15;
  const int hi   = lane >> 4;                            // sub-lane half
  const float* xrow = x + (m0 + n) * NFEAT;              // lanes L, L+16 share row

  // ---- pass 1: per-row max (two lanes split the row, xor-16 combine) ----
  float mx = -3.402823466e38f;
  const float* xh = xrow + hi * (NFEAT / 2);
#pragma unroll 4
  for (int j = 0; j < NFEAT / 8; ++j) {
    v4f v = *(const v4f*)(xh + (j << 2));
    mx = fmaxf(mx, fmaxf(fmaxf(v.x, v.y), fmaxf(v.z, v.w)));
  }
  mx = fmaxf(mx, swz_xor<16>(mx));
  const float mx2 = mx * LOG2E;

  // ---- pass 2: double-buffered K loop ----
  v8f acc[4];
#pragma unroll
  for (int t = 0; t < 4; ++t)
#pragma unroll
    for (int i = 0; i < 8; ++i) acc[t][i] = 0.f;

  const int koff = hi ? 16 : 0;                   // A fragment sub-lane K offset
  const int kB   = hi ? 16 : 0;                   // B fragment sub-lane K offset
  const _Float16* pbase = p + (nh + n) * NFEAT + kB;

  v16h e0, e1, e2, e3;                            // even-chunk B buffer
  v16h o0, o1, o2, o3;                            // odd-chunk B buffer
  LDB4(e0, e1, e2, e3, 0);
  XQ xe = ldx(xrow + koff);

  for (int kc = 0; kc < NFEAT; kc += 64) {
    // prefetch odd chunk while even WMMAs run
    const int k1 = kc + 32;
    LDB4(o0, o1, o2, o3, k1);
    XQ xo = ldx(xrow + k1 + koff);
    SCHED_FENCE();
    {
      v16h a = expa(xe, mx2);
      acc[0] = WMMA16(a, e0, acc[0]);
      acc[1] = WMMA16(a, e1, acc[1]);
      acc[2] = WMMA16(a, e2, acc[2]);
      acc[3] = WMMA16(a, e3, acc[3]);
    }
    SCHED_FENCE();
    // prefetch next even chunk (wrapped in-bounds on final iteration; values unused)
    const int k2 = (kc + 64) & (NFEAT - 1);
    LDB4(e0, e1, e2, e3, k2);
    xe = ldx(xrow + k2 + koff);
    SCHED_FENCE();
    {
      v16h a = expa(xo, mx2);
      acc[0] = WMMA16(a, o0, acc[0]);
      acc[1] = WMMA16(a, o1, acc[1]);
      acc[2] = WMMA16(a, o2, acc[2]);
      acc[3] = WMMA16(a, o3, acc[3]);
    }
    SCHED_FENCE();
  }

  // ---- epilogue: out = mx_row + ln(S). C layout: VGPR j holds M = j + 8*hi ----
#pragma unroll
  for (int j = 0; j < 8; ++j) {
    const float rmx = bcast_lane(mx, j + (hi << 3));
    float* orow = out + (m0 + (hi << 3) + j) * NCLS + nh + n;
#pragma unroll
    for (int t = 0; t < 4; ++t)
      orow[t * 16] = __builtin_fmaf(LN2, hw_log2(acc[t][j]), rmx);
  }
}

extern "C" void kernel_launch(void* const* d_in, const int* in_sizes, int n_in,
                              void* d_out, int out_size, void* d_ws, size_t ws_size,
                              hipStream_t stream) {
  (void)in_sizes; (void)n_in; (void)out_size; (void)ws_size;
  const float* x = (const float*)d_in[0];        // (2048, 16, 64) f32, row-major flat
  const float* w = (const float*)d_in[1];        // (128, 1024) f32
  float* out = (float*)d_out;                    // (2048, 128) f32
  _Float16* p = (_Float16*)d_ws;                 // 128*1024 f16 = 256 KB scratch

  // 1) softmax(weight) -> f16 probabilities (one wave per class row)
  wsoftmax_f16_kernel<<<NCLS / 4, 128, 0, stream>>>(w, p);
  // 2) fused exp + WMMA GEMM + log epilogue: 256 waves, 16 rows x 64 cols each
  lse_wmma_kernel<<<(NBATCH / 16) * 2 / 4, 128, 0, stream>>>(x, p, out);
}
